// GraphLabelEncoder_12120397709738
// MI455X (gfx1250) — compile-verified, matrix operand order checked by reading, not run on hardware
//
#include <hip/hip_runtime.h>
#include <hip/hip_bf16.h>
#include <hip/hip_fp16.h>

#define D 256
#define LDSPITCH 264  // 256 + 8 halfs pad: 528B row pitch, 16B-aligned, bank-rotating

typedef __attribute__((ext_vector_type(16))) _Float16 v16h;
typedef __attribute__((ext_vector_type(8)))  float    v8f;
typedef uint u32x4 __attribute__((ext_vector_type(4)));
typedef uint u32x8 __attribute__((ext_vector_type(8)));

union Frag16 { uint4 q[2]; v16h v; };

__device__ __forceinline__ float gelu_exact(float x) {
    return 0.5f * x * (1.0f + erff(x * 0.70710678118654752f));
}

// ---------------- elementwise converts ----------------
__global__ __launch_bounds__(256) void k_f32_to_f16(const float* __restrict__ in,
                                                    _Float16* __restrict__ out, int n) {
    int i = blockIdx.x * 256 + threadIdx.x;
    if (i < n) out[i] = (_Float16)in[i];
}

__global__ __launch_bounds__(256) void k_gelu_f16(const float* __restrict__ in,
                                                  _Float16* __restrict__ out, int n) {
    int i = blockIdx.x * 256 + threadIdx.x;
    if (i < n) out[i] = (_Float16)gelu_exact(in[i]);
}

// ---------------- WMMA GEMM core: 2 A-stripes per wave, shared B fragment -------------
template <bool FULL>
__device__ __forceinline__ void gemm_body(const Frag16* a0, const Frag16* a1,
                                          const _Float16* lw, const float* lbias,
                                          float* __restrict__ out,
                                          int row0, int m, int hi, int nrows) {
#pragma unroll 1
    for (int jt = 0; jt < 16; ++jt) {       // 16 N-tiles of 16 columns
        const _Float16* brow = &lw[(jt * 16 + m) * LDSPITCH];  // B col j = W row j
        v8f acc0 = {}, acc1 = {};
#pragma unroll
        for (int kc = 0; kc < 8; ++kc) {
            Frag16 b;
            b.q[0] = *(const uint4*)&brow[kc * 32 + hi * 8];
            b.q[1] = *(const uint4*)&brow[kc * 32 + 16 + hi * 8];
            // Two independent accumulator chains share one B fragment:
            acc0 = __builtin_amdgcn_wmma_f32_16x16x32_f16(
                false, a0[kc].v, false, b.v, (short)0, acc0, false, false);
            acc1 = __builtin_amdgcn_wmma_f32_16x16x32_f16(
                false, a1[kc].v, false, b.v, (short)0, acc1, false, false);
        }
        float bc = lbias[jt * 16 + m];      // ds_load, no v_movrels register indexing
        // C/D layout: lane -> N = m; VGPR r -> M = r + 8*hi
#pragma unroll
        for (int r = 0; r < 8; ++r) {
            int gw0 = row0 + r + 8 * hi;
            int gw1 = gw0 + 16;
            if (FULL || gw0 < nrows) out[(size_t)gw0 * D + jt * 16 + m] = acc0[r] + bc;
            if (FULL || gw1 < nrows) out[(size_t)gw1 * D + jt * 16 + m] = acc1[r] + bc;
        }
    }
}

// out[n,j] = sum_k A[n,k]*W[j,k] + bias[j].  A:[nrows,256] f16, W:[256,256] f16.
// Block: 256 thr = 8 waves; each wave computes 32 rows x 256 cols (256 rows/block).
// W tile is DMA'd into padded LDS by the Tensor Data Mover (one instruction).
__global__ __launch_bounds__(256) void k_gemm_wmma(const _Float16* __restrict__ A,
                                                   const _Float16* __restrict__ W,
                                                   const float* __restrict__ bias,
                                                   float* __restrict__ out, int nrows) {
    __shared__ _Float16 lw[D * LDSPITCH];  // 132 KB: whole weight matrix, padded pitch
    __shared__ float lbias[D];             // 1 KB bias

    int tid = threadIdx.x;
    lbias[tid] = bias[tid];
    if (tid < 32) {
        // Tensor DMA descriptor (ISA ch.8): 2D 256x256 tile of 2-byte elems,
        // LDS padding: 4 DWORDs (8 halfs) after every 128 DWORDs (one 256-half row)
        // -> reproduces LDSPITCH=264 exactly.
        uint ldsoff = (uint)(size_t)(void*)lw;        // low 32b of flat addr = LDS offset
        unsigned long long ga = (unsigned long long)(size_t)W;
        u32x4 g0;
        g0[0] = 0x1u;                                  // count=1, user mode, no gather
        g0[1] = ldsoff;                                // lds_addr
        g0[2] = (uint)ga;                              // global_addr[31:0]
        g0[3] = (uint)((ga >> 32) & 0x01FFFFFFu) | 0x80000000u;  // addr[56:32] | type=2
        u32x8 g1;
        g1[0] = (1u << 16)      // data_size = 1 (2 bytes)
              | (1u << 20)      // pad_enable
              | (6u << 22)      // pad_interval: 128 DWORDs
              | (3u << 25);     // pad_amount: 4 DWORDs
        g1[1] = 256u << 16;     // tensor_dim0[15:0] in bits [63:48]
        g1[2] = 256u << 16;     // tensor_dim1[15:0] in bits [95:80]
        g1[3] = 256u << 16;     // tile_dim0 in bits [127:112]
        g1[4] = 256u;           // tile_dim1 in bits [143:128]
        g1[5] = 256u;           // tensor_dim0_stride[31:0]
        g1[6] = 0u;
        g1[7] = 0u;
        asm volatile("tensor_load_to_lds %0, %1" :: "s"(g0), "s"(g1) : "memory");
        __builtin_amdgcn_s_wait_tensorcnt(0);
    }
    __syncthreads();

    int wave = tid >> 5, lane = tid & 31;
    int m = lane & 15, hi = lane >> 4;
    int row0 = blockIdx.x * 256 + wave * 32;
    if (row0 >= nrows) return;

    // A fragments: ISA 16-bit A 16x32 layout per stripe:
    //   VGPR0..3 <- K = hi*8 .. hi*8+7 ; VGPR4..7 <- K = 16+hi*8 .. 16+hi*8+7
    int gr0 = row0 + m;
    int gr1 = row0 + 16 + m;
    int cl = nrows - 1;
    if (gr0 > cl) gr0 = cl;                 // clamp: EXEC must stay all-ones for WMMA
    if (gr1 > cl) gr1 = cl;
    const uint4* xr0 = (const uint4*)(A + (size_t)gr0 * D);  // 32 uint4 per row
    const uint4* xr1 = (const uint4*)(A + (size_t)gr1 * D);
    Frag16 a0[8], a1[8];
#pragma unroll
    for (int kc = 0; kc < 8; ++kc) {        // K chunk kc covers k = kc*32..kc*32+31
        a0[kc].q[0] = xr0[kc * 4 + hi];
        a0[kc].q[1] = xr0[kc * 4 + 2 + hi];
        a1[kc].q[0] = xr1[kc * 4 + hi];
        a1[kc].q[1] = xr1[kc * 4 + 2 + hi];
    }

    if (row0 + 32 <= nrows)
        gemm_body<true>(a0, a1, lw, lbias, out, row0, m, hi, nrows);   // branch-free
    else
        gemm_body<false>(a0, a1, lw, lbias, out, row0, m, hi, nrows);  // ragged tail
}

// ---------------- COO SPMM: out[row] += val * h[col], one wave per edge ----------------
__global__ __launch_bounds__(256) void k_spmm(const long long* __restrict__ rows,
                                              const long long* __restrict__ cols,
                                              const float* __restrict__ vals,
                                              const float* __restrict__ h,
                                              float* __restrict__ out, int E) {
    int wave = threadIdx.x >> 5, lane = threadIdx.x & 31;
    int e = blockIdx.x * 8 + wave;
    if (e >= E) return;
    int r = (int)rows[e], c = (int)cols[e];
    float v = vals[e];
    const float4* src = (const float4*)(h + (size_t)c * D);
    float* dst = out + (size_t)r * D;
    float4 p0 = src[lane];
    float4 p1 = src[lane + 32];
    int b0 = lane * 4;
    atomicAdd(dst + b0 + 0, v * p0.x);
    atomicAdd(dst + b0 + 1, v * p0.y);
    atomicAdd(dst + b0 + 2, v * p0.z);
    atomicAdd(dst + b0 + 3, v * p0.w);
    atomicAdd(dst + 128 + b0 + 0, v * p1.x);
    atomicAdd(dst + 128 + b0 + 1, v * p1.y);
    atomicAdd(dst + 128 + b0 + 2, v * p1.z);
    atomicAdd(dst + 128 + b0 + 3, v * p1.w);
}

// ---------------- GELU + residual + LayerNorm, one block per row ----------------
__global__ __launch_bounds__(256) void k_final(const float* __restrict__ h,
                                               const float* __restrict__ x,
                                               const float* __restrict__ gamma,
                                               const float* __restrict__ beta,
                                               float* __restrict__ out, int nrows) {
    int row = blockIdx.x, tid = threadIdx.x;
    size_t base = (size_t)row * D + tid;
    float y = gelu_exact(h[base]) + x[base];

    float s = y, s2 = y * y;
#pragma unroll
    for (int off = 16; off > 0; off >>= 1) {
        s  += __shfl_down(s,  off, 32);
        s2 += __shfl_down(s2, off, 32);
    }
    __shared__ float ps[8], ps2[8];
    int wave = tid >> 5, lane = tid & 31;
    if (lane == 0) { ps[wave] = s; ps2[wave] = s2; }
    __syncthreads();
    float sum = 0.f, sum2 = 0.f;
#pragma unroll
    for (int i = 0; i < 8; ++i) { sum += ps[i]; sum2 += ps2[i]; }

    float mu  = sum * (1.0f / 256.0f);
    float var = sum2 * (1.0f / 256.0f) - mu * mu;
    float inv = rsqrtf(var + 1e-5f);
    out[base] = (y - mu) * inv * gamma[tid] + beta[tid];
}

extern "C" void kernel_launch(void* const* d_in, const int* in_sizes, int n_in,
                              void* d_out, int out_size, void* d_ws, size_t ws_size,
                              hipStream_t stream) {
    const float* x     = (const float*)d_in[0];
    const float* W1    = (const float*)d_in[1];
    const float* b1    = (const float*)d_in[2];
    const float* W2    = (const float*)d_in[3];
    const float* b2    = (const float*)d_in[4];
    const float* gamma = (const float*)d_in[5];
    const float* beta  = (const float*)d_in[6];
    const float* vals  = (const float*)d_in[7];
    const long long* idx = (const long long*)d_in[8];

    int N = in_sizes[0] / D;
    int E = in_sizes[7];
    const long long* rows = idx;
    const long long* cols = idx + E;

    size_t nd = (size_t)N * D;
    char* ws = (char*)d_ws;
    _Float16* xh  = (_Float16*)ws; ws += nd * 2;
    _Float16* gh  = (_Float16*)ws; ws += nd * 2;
    _Float16* w1h = (_Float16*)ws; ws += (size_t)D * D * 2;
    _Float16* w2h = (_Float16*)ws; ws += (size_t)D * D * 2;
    float* hA = (float*)ws; ws += nd * 4;   // GEMM outputs
    float* hB = (float*)ws; ws += nd * 4;   // SPMM accumulators

    int ndi = (int)nd;
    int eltGrid = (ndi + 255) / 256;
    int gemmGrid = (N + 255) / 256;
    int spmmGrid = (E + 7) / 8;

    // One-time converts to f16
    k_f32_to_f16<<<(D * D + 255) / 256, 256, 0, stream>>>(W1, w1h, D * D);
    k_f32_to_f16<<<(D * D + 255) / 256, 256, 0, stream>>>(W2, w2h, D * D);
    k_f32_to_f16<<<eltGrid, 256, 0, stream>>>(x, xh, ndi);

    // Layer 1: GEMM -> SPMM -> GELU(+f16)
    k_gemm_wmma<<<gemmGrid, 256, 0, stream>>>(xh, w1h, b1, hA, N);
    hipMemsetAsync(hB, 0, nd * 4, stream);
    k_spmm<<<spmmGrid, 256, 0, stream>>>(rows, cols, vals, hA, hB, E);
    k_gelu_f16<<<eltGrid, 256, 0, stream>>>(hB, gh, ndi);

    // Layer 2: GEMM -> SPMM
    k_gemm_wmma<<<gemmGrid, 256, 0, stream>>>(gh, w2h, b2, hA, N);
    hipMemsetAsync(hB, 0, nd * 4, stream);
    k_spmm<<<spmmGrid, 256, 0, stream>>>(rows, cols, vals, hA, hB, E);

    // GELU + residual + LayerNorm
    k_final<<<N, 256, 0, stream>>>(hB, x, gamma, beta, (float*)d_out, N);
}
